// HaloAttn_89455578841458
// MI455X (gfx1250) — compile-verified, matrix operand order checked by reading, not run on hardware
//
#include <hip/hip_runtime.h>
#include <hip/hip_bf16.h>
#include <math.h>

typedef __attribute__((ext_vector_type(16))) __bf16 v16bf;
typedef __attribute__((ext_vector_type(8)))  float  v8f;

#define B_     4
#define H_     128
#define W_     128
#define C_     256
#define NHEADS 8
#define DH     32
#define BLOCK_ 8
#define HALO_  3
#define WIN_   14
#define NB     256         // 16x16 blocks
#define QWIN   64          // 8*8
#define KVWIN  196         // 14*14
#define KVPAD  224         // padded to multiple of 32
#define NPIX   (B_*H_*W_)  // 65536
#define QKV_LD 768
#define SCALE_ 0.17677669529663687f  // 32^-0.5

__device__ __forceinline__ v8f wmma_bf16(v16bf a, v16bf b, v8f c) {
  return __builtin_amdgcn_wmma_f32_16x16x32_bf16(false, a, false, b,
                                                 (short)0, c, false, false);
}

// 16-bit A/B tile fragment for one lane: K elements at [khalf*8 + 0..7] and
// [16 + khalf*8 + 0..7]; caller passes p already offset by (row*ld + k0 + khalf*8).
__device__ __forceinline__ v16bf load_tile_bf16(const __bf16* p) {
  v16bf r;
#pragma unroll
  for (int i = 0; i < 8; ++i) r[i] = p[i];
#pragma unroll
  for (int i = 0; i < 8; ++i) r[8 + i] = p[16 + i];
  return r;
}

__device__ __forceinline__ v16bf load_tile_f32(const float* p) {
  v16bf r;
#pragma unroll
  for (int i = 0; i < 8; ++i) r[i] = (__bf16)p[i];
#pragma unroll
  for (int i = 0; i < 8; ++i) r[8 + i] = (__bf16)p[16 + i];
  return r;
}

// ---------------- prep: transpose + bf16-convert weights -------------------
__global__ void prep_weights(const float* __restrict__ Wq, const float* __restrict__ Wk,
                             const float* __restrict__ Wv, const float* __restrict__ Wp,
                             __bf16* __restrict__ WqkvT, __bf16* __restrict__ WprojT) {
  int t = blockIdx.x * blockDim.x + threadIdx.x;
  if (t < 768 * 256) {                 // WqkvT[n][k] = W*[k][n&255]
    int n = t >> 8, k = t & 255;
    const float* src = (n < 256) ? Wq : (n < 512 ? Wk : Wv);
    WqkvT[t] = (__bf16)src[k * 256 + (n & 255)];
  } else {
    int u = t - 768 * 256;
    if (u < 256 * 256) {
      int n = u >> 8, k = u & 255;
      WprojT[u] = (__bf16)Wp[k * 256 + n];
    }
  }
}

// ---------------- prep: projected positional embeddings --------------------
__device__ __forceinline__ float pos_chan(int jy, int jx, int c) {
  float coord; int cc;
  if (c < 128) { coord = (float)(jy + 1); cc = c; }
  else         { coord = (float)(jx + 1); cc = c - 128; }
  const float twopi = 6.28318530717958647692f;
  float e = coord / (14.0f + 1e-6f) * twopi;
  int pair = cc >> 1;
  float t = __powf(10000.0f, (float)(2 * pair) * (1.0f / 128.0f));
  float p = e / t;
  return (cc & 1) ? __cosf(p) : __sinf(p);
}

__global__ void prep_pos(const float* __restrict__ Wq, const float* __restrict__ Wk,
                         __bf16* __restrict__ qpos, __bf16* __restrict__ kpos) {
  int t = blockIdx.x * blockDim.x + threadIdx.x;   // (QWIN + KVPAD) * 256 threads
  int cout = t & 255;
  int row  = t >> 8;
  if (row < QWIN) {                                 // q_pos @ Wq
    int my = row >> 3, mx = row & 7;
    float acc = 0.f;
    for (int c = 0; c < 256; ++c)
      acc += pos_chan(my + HALO_, mx + HALO_, c) * Wq[c * 256 + cout];
    qpos[row * 256 + cout] = (__bf16)acc;
  } else {
    int j = row - QWIN;
    if (j < KVWIN) {                                // k_pos @ Wk
      int jy = j / WIN_, jx = j % WIN_;
      float acc = 0.f;
      for (int c = 0; c < 256; ++c)
        acc += pos_chan(jy, jx, c) * Wk[c * 256 + cout];
      kpos[j * 256 + cout] = (__bf16)acc;
    } else if (j < KVPAD) {
      kpos[j * 256 + cout] = (__bf16)0.0f;          // zero pad rows
    }
  }
}

// ---------------- big QKV GEMM: [65536 x 256] @ [256 x 768] ----------------
__global__ void __launch_bounds__(256) qkv_gemm(const float* __restrict__ x,
                                                const __bf16* __restrict__ WqkvT,
                                                __bf16* __restrict__ qkv) {
  int wave = threadIdx.x >> 5, lane = threadIdx.x & 31;
  int tile = blockIdx.x * 8 + wave;   // 196608 tiles total
  int mt = tile / 48, nt = tile % 48;
  int l16 = lane & 15, kh = lane >> 4;
  const float*  arow = x      + (size_t)(mt * 16 + l16) * 256 + kh * 8;
  const __bf16* brow = WqkvT  + (size_t)(nt * 16 + l16) * 256 + kh * 8;
  v8f acc = {};
#pragma unroll
  for (int kt = 0; kt < 8; ++kt) {
    v16bf a = load_tile_f32(arow + kt * 32);
    v16bf b = load_tile_bf16(brow + kt * 32);
    acc = wmma_bf16(a, b, acc);
  }
  int col = nt * 16 + l16;
  size_t base = (size_t)(mt * 16 + kh * 8);
#pragma unroll
  for (int r = 0; r < 8; ++r)
    qkv[(base + r) * QKV_LD + col] = (__bf16)acc[r];
}

// ---------------- halo attention per (batch, block, head) ------------------
__global__ void __launch_bounds__(128) halo_attn(
    const __bf16* __restrict__ qkv, const __bf16* __restrict__ qpos,
    const __bf16* __restrict__ kpos, __bf16* __restrict__ aout) {
  __shared__ __bf16 ks[KVPAD][DH];      // K + kpos, row-major          14336 B
  __shared__ __bf16 vT[DH][KVPAD];      // V transposed (B-tile reads)  14336 B
  __shared__ __bf16 pb[4][16][KVPAD];   // probs, per wave              28672 B

  int gid  = blockIdx.x;
  int head = gid & 7;
  int blk  = (gid >> 3) & 255;
  int b    = gid >> 11;
  int by = blk >> 4, bx = blk & 15;
  int tid = threadIdx.x;
  int wave = tid >> 5, lane = tid & 31;
  int l16 = lane & 15, kh = lane >> 4;

  // cooperative K(+pos)/V staging into LDS
  for (int j = tid; j < KVPAD; j += 128) {
    int jy = j / WIN_, jx = j - jy * WIN_;
    int gy = by * BLOCK_ - HALO_ + jy, gx = bx * BLOCK_ - HALO_ + jx;
    bool ok = (j < KVWIN) && ((unsigned)gy < H_) && ((unsigned)gx < W_);
    if (ok) {
      size_t pix = ((size_t)b * H_ + gy) * W_ + gx;
      const __bf16* kg = qkv + pix * QKV_LD + 256 + head * DH;
      const __bf16* vg = qkv + pix * QKV_LD + 512 + head * DH;
      const __bf16* kp = kpos + (size_t)j * 256 + head * DH;
#pragma unroll
      for (int d = 0; d < DH; ++d) {
        ks[j][d] = (__bf16)((float)kg[d] + (float)kp[d]);
        vT[d][j] = vg[d];
      }
    } else {
#pragma unroll
      for (int d = 0; d < DH; ++d) { ks[j][d] = (__bf16)0.f; vT[d][j] = (__bf16)0.f; }
    }
  }
  __syncthreads();

  // Q A-tile (16 rows x K=32) for this wave, with qpos added
  int m  = wave * 16 + l16;
  int qy = by * BLOCK_ + (m >> 3), qx = bx * BLOCK_ + (m & 7);
  size_t qpix = ((size_t)b * H_ + qy) * W_ + qx;
  const __bf16* qg = qkv  + qpix * QKV_LD + head * DH + kh * 8;
  const __bf16* qp = qpos + (size_t)m * 256 + head * DH + kh * 8;
  v16bf aq;
#pragma unroll
  for (int i = 0; i < 8; ++i) aq[i]     = (__bf16)((float)qg[i]      + (float)qp[i]);
#pragma unroll
  for (int i = 0; i < 8; ++i) aq[8 + i] = (__bf16)((float)qg[16 + i] + (float)qp[16 + i]);

  // scores: 14 WMMA tiles of [16 x 16], K=32
  v8f s[14];
#pragma unroll
  for (int nt = 0; nt < 14; ++nt) {
    v16bf bk = load_tile_bf16(&ks[nt * 16 + l16][kh * 8]);
    v8f c = {};
    c = wmma_bf16(aq, bk, c);
    int j  = nt * 16 + l16;
    int jy = j / WIN_, jx = j - jy * WIN_;
    int gy = by * BLOCK_ - HALO_ + jy, gx = bx * BLOCK_ - HALO_ + jx;
    bool ok = (j < KVWIN) && ((unsigned)gy < H_) && ((unsigned)gx < W_);
#pragma unroll
    for (int r = 0; r < 8; ++r)
      s[nt][r] = ok ? c[r] * SCALE_ : -3.0e38f;
  }

  // softmax: C-layout => VGPR r holds row r (lanes 0-15) / r+8 (lanes 16-31);
  // per-row reduce over the 16-lane half via xor shuffles.
#pragma unroll
  for (int r = 0; r < 8; ++r) {
    float mx = -3.0e38f;
#pragma unroll
    for (int nt = 0; nt < 14; ++nt) mx = fmaxf(mx, s[nt][r]);
    mx = fmaxf(mx, __shfl_xor(mx, 1, 32));
    mx = fmaxf(mx, __shfl_xor(mx, 2, 32));
    mx = fmaxf(mx, __shfl_xor(mx, 4, 32));
    mx = fmaxf(mx, __shfl_xor(mx, 8, 32));
    float sum = 0.f;
#pragma unroll
    for (int nt = 0; nt < 14; ++nt) { float e = __expf(s[nt][r] - mx); s[nt][r] = e; sum += e; }
    sum += __shfl_xor(sum, 1, 32);
    sum += __shfl_xor(sum, 2, 32);
    sum += __shfl_xor(sum, 4, 32);
    sum += __shfl_xor(sum, 8, 32);
    float inv = 1.0f / sum;
#pragma unroll
    for (int nt = 0; nt < 14; ++nt) s[nt][r] *= inv;
  }

  // probs to LDS (re-layout C-format -> A-format via memory)
#pragma unroll
  for (int nt = 0; nt < 14; ++nt) {
    int j = nt * 16 + l16;
#pragma unroll
    for (int r = 0; r < 8; ++r)
      pb[wave][r + kh * 8][j] = (__bf16)s[nt][r];
  }
  __syncthreads();

  // out = P[16x224] @ V[224x32] : 2 N-tiles x 7 K-steps
#pragma unroll
  for (int ntile = 0; ntile < 2; ++ntile) {
    v8f acc = {};
#pragma unroll
    for (int kt = 0; kt < 7; ++kt) {
      v16bf pa = load_tile_bf16(&pb[wave][l16][kt * 32 + kh * 8]);
      v16bf vb = load_tile_bf16(&vT[ntile * 16 + l16][kt * 32 + kh * 8]);
      acc = wmma_bf16(pa, vb, acc);
    }
    int ch = head * DH + ntile * 16 + l16;
#pragma unroll
    for (int r = 0; r < 8; ++r) {
      int mr = wave * 16 + kh * 8 + r;
      int oy = by * BLOCK_ + (mr >> 3), ox = bx * BLOCK_ + (mr & 7);
      size_t opix = ((size_t)b * H_ + oy) * W_ + ox;
      aout[opix * C_ + ch] = (__bf16)acc[r];
    }
  }
}

// ---------------- output projection: [65536 x 256] @ [256 x 256] + b -------
__global__ void __launch_bounds__(256) out_proj(const __bf16* __restrict__ ain,
                                                const __bf16* __restrict__ WprojT,
                                                const float* __restrict__ bias,
                                                float* __restrict__ out) {
  int wave = threadIdx.x >> 5, lane = threadIdx.x & 31;
  int tile = blockIdx.x * 8 + wave;   // 65536 tiles total
  int mt = tile >> 4, nt = tile & 15;
  int l16 = lane & 15, kh = lane >> 4;
  const __bf16* arow = ain    + (size_t)(mt * 16 + l16) * 256 + kh * 8;
  const __bf16* brow = WprojT + (size_t)(nt * 16 + l16) * 256 + kh * 8;
  v8f acc = {};
#pragma unroll
  for (int kt = 0; kt < 8; ++kt)
    acc = wmma_bf16(load_tile_bf16(arow + kt * 32), load_tile_bf16(brow + kt * 32), acc);
  int col = nt * 16 + l16;
  float bv = bias[col];
  size_t base = (size_t)(mt * 16 + kh * 8);
#pragma unroll
  for (int r = 0; r < 8; ++r)
    out[(base + r) * 256 + col] = acc[r] + bv;
}

extern "C" void kernel_launch(void* const* d_in, const int* in_sizes, int n_in,
                              void* d_out, int out_size, void* d_ws, size_t ws_size,
                              hipStream_t stream) {
  const float* x  = (const float*)d_in[0];
  const float* Wq = (const float*)d_in[1];
  const float* Wk = (const float*)d_in[2];
  const float* Wv = (const float*)d_in[3];
  const float* Wp = (const float*)d_in[4];
  const float* bp = (const float*)d_in[5];
  float* out = (float*)d_out;

  char* ws = (char*)d_ws;
  __bf16* qkv    = (__bf16*)ws;  ws += (size_t)NPIX * QKV_LD * 2;  // 100.7 MB
  __bf16* aout   = (__bf16*)ws;  ws += (size_t)NPIX * C_ * 2;      //  33.6 MB
  __bf16* WqkvT  = (__bf16*)ws;  ws += (size_t)768 * 256 * 2;
  __bf16* WprojT = (__bf16*)ws;  ws += (size_t)256 * 256 * 2;
  __bf16* qpos   = (__bf16*)ws;  ws += (size_t)QWIN * 256 * 2;
  __bf16* kpos   = (__bf16*)ws;  ws += (size_t)KVPAD * 256 * 2;
  (void)ws; (void)ws_size; (void)in_sizes; (void)n_in; (void)out_size;

  prep_weights<<<(768 * 256 + 256 * 256) / 256, 256, 0, stream>>>(Wq, Wk, Wv, Wp, WqkvT, WprojT);
  prep_pos<<<(QWIN + KVPAD), 256, 0, stream>>>(Wq, Wk, qpos, kpos);
  qkv_gemm<<<(4096 * 48) / 8, 256, 0, stream>>>(x, WqkvT, qkv);
  halo_attn<<<B_ * NB * NHEADS, 128, 0, stream>>>(qkv, qpos, kpos, aout);
  out_proj<<<(4096 * 16) / 8, 256, 0, stream>>>(aout, WprojT, bp, out);
}
